// BinaryMatchAttention_62380105007277
// MI455X (gfx1250) — matrix operand order repaired and checked by reading, not compiled, not run on hardware
//
#include <hip/hip_runtime.h>
#include <math.h>

typedef __attribute__((ext_vector_type(16))) _Float16 v16h;
typedef __attribute__((ext_vector_type(8)))  float    v8f;

constexpr int BB = 4;
constexpr int QQ = 1024;
constexpr int KK = 2048;
constexpr int VV = 512;
constexpr int NB = 16;          // NUM_BITS
constexpr float EPSV = 1e-8f;
// scores = sum ln(match)/T = (sum log2)*ln2/T, T=0.1
constexpr float LN2_OVER_T = 6.93147180559945f;   // ln2 * 10
constexpr float INV_LN2    = 1.4426950408889634f; // for exp(x)=exp2(x*INV_LN2)

__device__ __forceinline__ float fast_log2(float x) { return __builtin_amdgcn_logf(x); }   // v_log_f32
__device__ __forceinline__ float fast_exp2(float x) { return __builtin_amdgcn_exp2f(x); }  // v_exp_f32

__device__ __forceinline__ unsigned short f2h_bits(float x) {
    union { _Float16 h; unsigned short u; } cv;
    cv.h = (_Float16)x;
    return cv.u;
}

// ---------------------------------------------------------------------------
// Kernel 1: scores + masked softmax -> weights (fp32, written to d_out tail)
// one block per (b,q) row; 256 threads; 8 K-elements per thread
// ---------------------------------------------------------------------------
__global__ __launch_bounds__(256) void bma_scores_softmax(
    const float* __restrict__ qbits,   // (B,Q,16)
    const float* __restrict__ kbits,   // (B,K,16)
    const int*   __restrict__ mask,    // (B,K)
    float*       __restrict__ wts)     // (B,Q,K)
{
    constexpr int KPT = KK / 256;      // 8
    const int bq  = blockIdx.x;
    const int b   = bq / QQ;
    const int tid = threadIdx.x;
    const int wid = tid >> 5, lane = tid & 31;

    __shared__ float qsh[NB];
    __shared__ float red[8];

    if (tid < NB) qsh[tid] = qbits[(size_t)bq * NB + tid];
    __syncthreads();

    float qv[NB];
#pragma unroll
    for (int n = 0; n < NB; ++n) qv[n] = qsh[n];

    float s[KPT];
#pragma unroll
    for (int j = 0; j < KPT; ++j) {
        const int k = tid + j * 256;
        const float4* kb4 = (const float4*)(kbits + ((size_t)b * KK + k) * NB);
        float4 k0 = kb4[0], k1 = kb4[1], k2 = kb4[2], k3 = kb4[3];
        // product of 4 bit-matches per group; group >= 1e-32 > fp32 denorm min
        float p0 = (1.0f - fabsf(qv[0]  - k0.x) + EPSV) * (1.0f - fabsf(qv[1]  - k0.y) + EPSV)
                 * (1.0f - fabsf(qv[2]  - k0.z) + EPSV) * (1.0f - fabsf(qv[3]  - k0.w) + EPSV);
        float p1 = (1.0f - fabsf(qv[4]  - k1.x) + EPSV) * (1.0f - fabsf(qv[5]  - k1.y) + EPSV)
                 * (1.0f - fabsf(qv[6]  - k1.z) + EPSV) * (1.0f - fabsf(qv[7]  - k1.w) + EPSV);
        float p2 = (1.0f - fabsf(qv[8]  - k2.x) + EPSV) * (1.0f - fabsf(qv[9]  - k2.y) + EPSV)
                 * (1.0f - fabsf(qv[10] - k2.z) + EPSV) * (1.0f - fabsf(qv[11] - k2.w) + EPSV);
        float p3 = (1.0f - fabsf(qv[12] - k3.x) + EPSV) * (1.0f - fabsf(qv[13] - k3.y) + EPSV)
                 * (1.0f - fabsf(qv[14] - k3.z) + EPSV) * (1.0f - fabsf(qv[15] - k3.w) + EPSV);
        float ls = fast_log2(p0) + fast_log2(p1) + fast_log2(p2) + fast_log2(p3);
        float sc = ls * LN2_OVER_T;
        s[j] = (mask[(size_t)b * KK + k] == 0) ? -INFINITY : sc;
    }

    // ---- block max ----
    float m = -INFINITY;
#pragma unroll
    for (int j = 0; j < KPT; ++j) m = fmaxf(m, s[j]);
#pragma unroll
    for (int off = 16; off > 0; off >>= 1) m = fmaxf(m, __shfl_xor(m, off, 32));
    if (lane == 0) red[wid] = m;
    __syncthreads();
    m = red[0];
#pragma unroll
    for (int w = 1; w < 8; ++w) m = fmaxf(m, red[w]);
    __syncthreads();

    // ---- exp + block sum ----
    float e[KPT];
    float sum = 0.0f;
#pragma unroll
    for (int j = 0; j < KPT; ++j) {
        e[j] = fast_exp2((s[j] - m) * INV_LN2);
        sum += e[j];
    }
#pragma unroll
    for (int off = 16; off > 0; off >>= 1) sum += __shfl_xor(sum, off, 32);
    if (lane == 0) red[wid] = sum;
    __syncthreads();
    float tot = red[0];
#pragma unroll
    for (int w = 1; w < 8; ++w) tot += red[w];
    const float inv = 1.0f / tot;

#pragma unroll
    for (int j = 0; j < KPT; ++j) {
        const int k = tid + j * 256;
        wts[(size_t)bq * KK + k] = e[j] * inv;
    }
}

// ---------------------------------------------------------------------------
// Kernel 2: output = weights @ values  (batched, f16 WMMA, f32 accumulate)
// workgroup tile 128(M) x 64(N), K-step 32; 8 waves, each wave 16x64
// ---------------------------------------------------------------------------
__global__ __launch_bounds__(256) void bma_pv_gemm(
    const float* __restrict__ wts,   // (B,Q,K) fp32
    const float* __restrict__ vals,  // (B,K,V) fp32
    float*       __restrict__ out)   // (B,Q,V) fp32
{
    constexpr int TM = 128, TN = 64, TK = 32;
    constexpr int LDW = 34;  // ushort stride (17 dwords, gcd(17,64)=1 -> conflict-free)

    __shared__ __align__(16) unsigned short ldsW[TM * LDW]; // A tile, row-major [m][k]
    __shared__ __align__(16) unsigned short ldsV[TN * LDW]; // B tile, [n][k] (k-major per column)
    unsigned int* ldsW32 = (unsigned int*)ldsW;
    unsigned int* ldsV32 = (unsigned int*)ldsV;

    const int tid  = threadIdx.x;
    const int wave = tid >> 5;
    const int lane = tid & 31;
    const int half = lane >> 4;
    const int l16  = lane & 15;

    const int b     = blockIdx.z;
    const int qbase = blockIdx.y * TM;
    const int nbase = blockIdx.x * TN;
    const int m0    = wave * 16;

    const float* Wp = wts  + ((size_t)b * QQ + qbase) * KK;
    const float* Vp = vals + (size_t)b * KK * VV + nbase;

    v8f acc[4];
#pragma unroll
    for (int i = 0; i < 4; ++i)
#pragma unroll
        for (int r = 0; r < 8; ++r) acc[i][r] = 0.0f;

    for (int kb = 0; kb < KK; kb += TK) {
        // prefetch next K-tile into caches (global_prefetch_b8)
        if (kb + TK < KK) {
            __builtin_prefetch(Wp + (size_t)(tid & 127) * KK + (kb + TK), 0, 1);
            __builtin_prefetch(Vp + (size_t)(kb + TK + (tid & 31)) * VV, 0, 1);
        }

        // stage W tile 128x32 fp32 -> fp16 (coalesced rows of 32)
#pragma unroll
        for (int e = 0; e < 16; ++e) {
            const int flat = tid + e * 256;
            const int r = flat >> 5, c = flat & 31;
            ldsW[r * LDW + c] = f2h_bits(Wp[(size_t)r * KK + kb + c]);
        }
        // stage V tile 32x64 fp32 -> fp16, transposed to [n][k]
#pragma unroll
        for (int e = 0; e < 8; ++e) {
            const int flat = tid + e * 256;
            const int kr = flat >> 6, nc = flat & 63;
            ldsV[nc * LDW + kr] = f2h_bits(Vp[(size_t)(kb + kr) * VV + nc]);
        }
        __syncthreads();

        // A fragment: 16x32 f16 per ISA layout:
        //   lanes 0-15 : M=lane,  elems = K{0..7,16..23}
        //   lanes 16-31: M=l16,   elems = K{8..15,24..31}
        union { unsigned int u[8]; v16h v; } a;
        const int arow = m0 + l16;
#pragma unroll
        for (int j = 0; j < 8; ++j) {
            const int cu = (j < 4 ? j : j + 4) + half * 4; // dword column in row
            a.u[j] = ldsW32[arow * (LDW / 2) + cu];
        }

        // 4 sub-tiles of N; reuse A fragment each time
#pragma unroll
        for (int nn = 0; nn < 4; ++nn) {
            // B fragment: 32x16 f16: lanes 0-15 col=lane K{0..7,16..23}, lanes 16-31 col=l16 K{8..15,24..31}
            union { unsigned int u[8]; v16h v; } bf;
            const int col = nn * 16 + l16;
#pragma unroll
            for (int j = 0; j < 8; ++j)
                bf.u[j] = ldsV32[col * (LDW / 2) + (j < 4 ? j : j + 4) + half * 4];

            acc[nn] = __builtin_amdgcn_wmma_f32_16x16x32_f16(
                false, a.v, false, bf.v, (short)0, acc[nn], false, false);
        }
        __syncthreads();
    }

    // D layout: VGPR r -> M = r + 8*half, N = l16
    float* Op = out + ((size_t)b * QQ + qbase + m0) * VV + nbase;
#pragma unroll
    for (int nn = 0; nn < 4; ++nn)
#pragma unroll
        for (int r = 0; r < 8; ++r)
            Op[(size_t)(r + 8 * half) * VV + nn * 16 + l16] = acc[nn][r];
}

// ---------------------------------------------------------------------------
extern "C" void kernel_launch(void* const* d_in, const int* in_sizes, int n_in,
                              void* d_out, int out_size, void* d_ws, size_t ws_size,
                              hipStream_t stream) {
    const float* qbits = (const float*)d_in[0];   // (B,Q,16)
    const float* kbits = (const float*)d_in[1];   // (B,K,16)
    const float* vals  = (const float*)d_in[2];   // (B,K,V)
    const int*   mask  = (const int*)d_in[3];     // (B,K)

    float* out = (float*)d_out;                      // (B,Q,V) first
    float* wts = out + (size_t)BB * QQ * VV;         // (B,Q,K) second

    // Phase 1: scores + softmax -> weights
    bma_scores_softmax<<<BB * QQ, 256, 0, stream>>>(qbits, kbits, mask, wts);

    // Phase 2: output = weights @ values (f16 WMMA, f32 accum)
    dim3 grid(VV / 64, QQ / 128, BB);
    bma_pv_gemm<<<grid, 256, 0, stream>>>(wts, vals, out);
}
